// low_light_transformer_66975720014330
// MI455X (gfx1250) — matrix operand order repaired and checked by reading, not compiled, not run on hardware
//
#include <hip/hip_runtime.h>
#include <hip/hip_bf16.h>
#include <math.h>

// ---------------------------------------------------------------------------
// Low-light transformer for MI455X (gfx1250, wave32, WMMA).
// Dense convs (3x3/1x1) are implicit-GEMM via v_wmma_f32_16x16x32_f16 with
// 4 accumulators per wave (16 Cout x 64 pixels), f32 accumulate.
// Weight K-tiles are staged block-wide through LDS with CDNA5 async-to-LDS
// loads (GLOBAL_LOAD_ASYNC_TO_LDS_B128 + s_wait_asynccnt), double-buffered
// against WMMA compute. Channel attention Gram + apply also run through WMMA.
// ---------------------------------------------------------------------------

#define USE_ASYNC_STAGE 1   // set 0 to fall back to synchronous LDS staging

typedef __attribute__((ext_vector_type(16))) _Float16 v16h;
typedef __attribute__((ext_vector_type(8)))  float    v8f;

#define NF    64
#define HEADS 4
#define HIDC  170   // int(64*2.66) = 170

#define ACT_NONE    0
#define ACT_LRELU   1
#define ACT_RELU    2
#define ACT_SIGMOID 3

// A-fragment K mapping for 16-bit 16x32 A matrix (ISA 7.12.2):
//   VGPR0..3: lanes0-15 K=0..7 (2/VGPR), lanes16-31 K=8..15
//   VGPR4..7: lanes0-15 K=16..23,        lanes16-31 K=24..31
__device__ __forceinline__ int kmapA(int lane, int i) {
  int klo = (lane & 16) ? 8 : 0;
  return (i < 8) ? (klo + i) : (16 + klo + (i - 8));
}
// B-fragment K mapping (K x 16, 16-bit), contiguous order:
//   lanes0-15 hold K=0..15 across the 8 VGPRs, lanes16-31 hold K=16..31
__device__ __forceinline__ int kmapB(int lane, int i) {
  return ((lane & 16) ? 16 : 0) + i;
}

// Stage one 1KB weight tile (32 lanes x 32B) global -> LDS.
// Issued by one wave; generic LDS pointer's low 32 bits are the LDS address
// (ISA 10.2).
__device__ __forceinline__ void stage_tile(const _Float16* gsrc,
                                           _Float16* lbuf, int lane) {
#if USE_ASYNC_STAGE
  unsigned int laddr = (unsigned int)(size_t)(lbuf + (size_t)lane * 16);
  unsigned long long ga = (unsigned long long)(size_t)(gsrc + (size_t)lane * 16);
  asm volatile("global_load_async_to_lds_b128 %0, %1, off"
               :: "v"(laddr), "v"(ga) : "memory");
  asm volatile("global_load_async_to_lds_b128 %0, %1, off offset:16"
               :: "v"(laddr), "v"(ga) : "memory");
#else
  v16h tmp = *(const v16h*)(gsrc + (size_t)lane * 16);
  *(v16h*)(lbuf + (size_t)lane * 16) = tmp;
#endif
}

__device__ __forceinline__ void stage_wait() {
#if USE_ASYNC_STAGE
  asm volatile("s_wait_asynccnt 0x0" ::: "memory");
#endif
}

// ---------------------------------------------------------------------------
// Weight packing: fp32 OIHW -> f16 A-fragments [ktile][mtile][lane][16].
// For 3x3 convs the packed K index is reordered to k = tap*Cin + ci so the
// conv kernel can decode (tap, ci) with shift/mask when Cin is a power of 2.
// ---------------------------------------------------------------------------
__global__ void __launch_bounds__(256) pack_w(const float* __restrict__ w,
                                              _Float16* __restrict__ pk,
                                              int Cout, int K, int Kp, int Mp,
                                              int Cin, int k3) {
  int tid = blockIdx.x * blockDim.x + threadIdx.x;
  int total = Kp * Mp * 32;
  if (tid >= total) return;
  int lane = tid & 31;
  int tm = tid >> 5;
  int m = tm % Mp;
  int t = tm / Mp;
  int co = m * 16 + (lane & 15);
  _Float16* dst = pk + (((size_t)t * Mp + m) * 32 + lane) * 16;
#pragma unroll
  for (int i = 0; i < 16; i++) {
    int k = t * 32 + kmapA(lane, i);
    float v = 0.f;
    if (co < Cout && k < K) {
      int src;
      if (k3) {
        int tap = k / Cin;       // tap = ky*3+kx
        int ci = k - tap * Cin;
        src = ci * 9 + tap;      // source OIHW flat index within [Cin*9]
      } else {
        src = k;
      }
      v = w[(size_t)co * K + src];
    }
    dst[i] = (_Float16)v;
  }
}

// ---------------------------------------------------------------------------
// Implicit-GEMM conv. One wave -> 16 Cout x 64 pixels (4 WMMA tiles).
// All 8 waves of a block share the same Cout tile (mt), so each weight K-tile
// is staged once per block into LDS (double-buffered, async) and A-fragments
// are broadcast-read from LDS.
// Requires: Wout power of two, Hout*Wout multiple of 64, Ng >= 8.
// ---------------------------------------------------------------------------
template <int K3>
__global__ void __launch_bounds__(256) conv_wmma(
    const float* __restrict__ in, int inCt, int inCo,
    const _Float16* __restrict__ wpk,
    const float* __restrict__ bias,
    const float* __restrict__ resid, int resCt, int resCo,
    float* __restrict__ out, int outCt, int outCo,
    int B, int Cin, int Cout, int Hin, int Win,
    int stride, int act, int Mp, int Kp,
    int WoutShift, int HWo, int gShift, int cinShift) {
  __shared__ _Float16 wtile[2][512];   // two 1KB weight tiles

  int wave = (blockIdx.x * blockDim.x + threadIdx.x) >> 5;
  int lane = threadIdx.x & 31;
  int wib = threadIdx.x >> 5;          // wave-in-block (0..7)
  int Ng = HWo >> 6;                   // 64-pixel groups (>= 64 for all calls)
  int ng = wave & (Ng - 1);
  int rest = wave >> gShift;
  int mt = rest % Mp;                  // same for all 8 waves in the block
  int b  = rest / Mp;

  int Wout = 1 << WoutShift;
  int HWi = Hin * Win;
  int K = K3 ? Cin * 9 : Cin;
  int ln = lane & 15;
  int khi = (lane & 16) ? 16 : 0;      // unused; B fragment uses kmapA order

  int nj[4], oyj[4], oxj[4];
#pragma unroll
  for (int j = 0; j < 4; j++) {
    int n = (ng << 6) + (j << 4) + ln;
    nj[j] = n;
    oyj[j] = (n >> WoutShift) * stride;   // pre-scaled by stride
    oxj[j] = (n & (Wout - 1)) * stride;
  }

  const float* inb = in + ((size_t)b * inCt + inCo) * HWi;

  // Prologue: stage tile 0
  if (wib == 0) {
    stage_tile(wpk + ((size_t)mt * 32) * 16, wtile[0], lane);
    stage_wait();
  }
  __syncthreads();

  v8f acc[4] = {};
  for (int t = 0; t < Kp; ++t) {
    // Kick off next tile while computing on the current one
    if (wib == 0 && t + 1 < Kp)
      stage_tile(wpk + (((size_t)(t + 1) * Mp + mt) * 32) * 16,
                 wtile[(t + 1) & 1], lane);

    v16h a = *(const v16h*)(&wtile[t & 1][(size_t)lane * 16]);

    v16h bfa[4];
    int kbase = (t << 5) + khi;
#pragma unroll
    for (int i = 0; i < 16; i++) {
      int k = (t << 5) + kmapA(lane, i);
      float vj[4] = {0.f, 0.f, 0.f, 0.f};
      if (k < K) {
        if (K3) {
          int tap, ci;
          if (cinShift >= 0) { ci = k & (Cin - 1); tap = k >> cinShift; }
          else               { tap = k / Cin; ci = k - tap * Cin; }
          int ky = tap / 3;
          int dx = tap - ky * 3 - 1;
          int dy = ky - 1;
          const float* rb = inb + (size_t)ci * HWi;
#pragma unroll
          for (int j = 0; j < 4; j++) {
            int iy = oyj[j] + dy;
            int ix = oxj[j] + dx;
            if ((unsigned)iy < (unsigned)Hin && (unsigned)ix < (unsigned)Win)
              vj[j] = rb[iy * Win + ix];
          }
        } else {
          const float* rb = inb + (size_t)k * HWi;
#pragma unroll
          for (int j = 0; j < 4; j++) vj[j] = rb[nj[j]];
        }
      }
#pragma unroll
      for (int j = 0; j < 4; j++) bfa[j][i] = (_Float16)vj[j];
    }
#pragma unroll
    for (int j = 0; j < 4; j++)
      acc[j] = __builtin_amdgcn_wmma_f32_16x16x32_f16(false, a, false, bfa[j],
                                                      (short)0, acc[j], false, false);

    if (t + 1 < Kp) {
      if (wib == 0) stage_wait();
      __syncthreads();
    }
  }

  int mrow = (lane & 16) ? 8 : 0;
#pragma unroll
  for (int j = 0; j < 4; j++) {
    int n = nj[j];
#pragma unroll
    for (int r = 0; r < 8; r++) {
      int co = (mt << 4) + mrow + r;
      if (co < Cout) {
        float v = acc[j][r];
        if (bias) v += bias[co];
        if (act == ACT_LRELU)        v = (v >= 0.f) ? v : 0.1f * v;
        else if (act == ACT_RELU)    v = fmaxf(v, 0.f);
        else if (act == ACT_SIGMOID) v = 1.f / (1.f + expf(-v));
        if (resid) v += resid[((size_t)b * resCt + resCo + co) * HWo + n];
        out[((size_t)b * outCt + outCo + co) * HWo + n] = v;
      }
    }
  }
}

// ---------------------------------------------------------------------------
// Depthwise 3x3 (pad 1, stride 1, no bias)
// ---------------------------------------------------------------------------
__global__ void __launch_bounds__(256) dwconv3x3(
    const float* __restrict__ in, int inCt, int inCo,
    const float* __restrict__ w,
    float* __restrict__ out, int outCt, int outCo,
    int B, int C, int H, int W) {
  int HW = H * W;
  int idx = blockIdx.x * blockDim.x + threadIdx.x;
  if (idx >= B * C * HW) return;
  int n = idx & (HW - 1);            // HW power of two
  int c = (idx / HW) % C;
  int b = idx / (HW * C);
  int y = n / W, x = n & (W - 1);
  const float* ib = in + ((size_t)b * inCt + inCo + c) * HW;
  const float* wc = w + c * 9;
  float s = 0.f;
#pragma unroll
  for (int ky = 0; ky < 3; ky++) {
    int iy = y + ky - 1;
    if ((unsigned)iy >= (unsigned)H) continue;
#pragma unroll
    for (int kx = 0; kx < 3; kx++) {
      int ix = x + kx - 1;
      if ((unsigned)ix >= (unsigned)W) continue;
      s += ib[iy * W + ix] * wc[ky * 3 + kx];
    }
  }
  out[((size_t)b * outCt + outCo + c) * HW + n] = s;
}

// ---------------------------------------------------------------------------
// Channel LayerNorm (mean/var over channel dim per pixel)
// ---------------------------------------------------------------------------
__global__ void __launch_bounds__(256) chan_ln_k(
    const float* __restrict__ in, int inCt, int inCo,
    const float* __restrict__ w, const float* __restrict__ bp,
    float* __restrict__ out, int outCt, int outCo,
    int B, int C, int HW) {
  int idx = blockIdx.x * blockDim.x + threadIdx.x;
  if (idx >= B * HW) return;
  int n = idx & (HW - 1);
  int b = idx / HW;
  const float* ib = in + ((size_t)b * inCt + inCo) * HW + n;
  float mu = 0.f;
  for (int c = 0; c < C; c++) mu += ib[(size_t)c * HW];
  mu /= (float)C;
  float var = 0.f;
  for (int c = 0; c < C; c++) { float d = ib[(size_t)c * HW] - mu; var += d * d; }
  var /= (float)C;
  float rs = rsqrtf(var + 1e-5f);
  float* ob = out + ((size_t)b * outCt + outCo) * HW + n;
  for (int c = 0; c < C; c++)
    ob[(size_t)c * HW] = (ib[(size_t)c * HW] - mu) * rs * w[c] + bp[c];
}

// ---------------------------------------------------------------------------
// L2 normalize rows of length HW (one block per (b,c) row), in place
// ---------------------------------------------------------------------------
__global__ void __launch_bounds__(256) l2norm_rows(
    float* __restrict__ buf, int Ct, int Co, int B, int C, int HW) {
  int row = blockIdx.x;
  int b = row / C, c = row % C;
  float* p = buf + ((size_t)b * Ct + Co + c) * HW;
  __shared__ float red[8];
  float s = 0.f;
  for (int i = threadIdx.x; i < HW; i += blockDim.x) { float v = p[i]; s += v * v; }
  for (int m = 16; m >= 1; m >>= 1) s += __shfl_xor(s, m, 32);
  int wv = threadIdx.x >> 5;
  if ((threadIdx.x & 31) == 0) red[wv] = s;
  __syncthreads();
  if (threadIdx.x == 0) {
    float t = 0.f;
    for (int i = 0; i < (int)(blockDim.x >> 5); i++) t += red[i];
    red[0] = t;
  }
  __syncthreads();
  float inv = 1.f / fmaxf(sqrtf(red[0]), 1e-12f);
  for (int i = threadIdx.x; i < HW; i += blockDim.x) p[i] *= inv;
}

// ---------------------------------------------------------------------------
// Attention Gram matrix S = softmax(temp * q k^T), one wave per (b,head).
// K reduction over HW via WMMA (2x unrolled); softmax in-wave via shfl_xor.
// ---------------------------------------------------------------------------
__global__ void __launch_bounds__(256) attn_gram(
    const float* __restrict__ qb, int qCt, int qCo,
    const float* __restrict__ kb, int kCt, int kCo,
    const float* __restrict__ temp,
    float* __restrict__ S, int B, int HW) {
  int wave = (blockIdx.x * blockDim.x + threadIdx.x) >> 5;
  int lane = threadIdx.x & 31;
  if (wave >= B * HEADS) return;
  int h = wave % HEADS, b = wave / HEADS;
  const float* qrow = qb + ((size_t)b * qCt + qCo + h * 16 + (lane & 15)) * HW;
  const float* krow = kb + ((size_t)b * kCt + kCo + h * 16 + (lane & 15)) * HW;
  int ka[16], kbb[16];
#pragma unroll
  for (int i = 0; i < 16; i++) { ka[i] = kmapA(lane, i); kbb[i] = kmapA(lane, i); }
  v8f acc = {};
  int Kp = HW >> 5;
  for (int t = 0; t < Kp; t += 2) {
    int base0 = t << 5;
    int base1 = base0 + 32;
    v16h a0, b0, a1, b1;
#pragma unroll
    for (int i = 0; i < 16; i++) {
      a0[i] = (_Float16)qrow[base0 + ka[i]];
      b0[i] = (_Float16)krow[base0 + kbb[i]];
      a1[i] = (_Float16)qrow[base1 + ka[i]];
      b1[i] = (_Float16)krow[base1 + kbb[i]];
    }
    acc = __builtin_amdgcn_wmma_f32_16x16x32_f16(false, a0, false, b0,
                                                 (short)0, acc, false, false);
    acc = __builtin_amdgcn_wmma_f32_16x16x32_f16(false, a1, false, b1,
                                                 (short)0, acc, false, false);
  }
  float tp = temp[h];
  int mrow = (lane & 16) ? 8 : 0;
#pragma unroll
  for (int r = 0; r < 8; r++) {
    float s = acc[r] * tp;
    float mx = s;
    for (int m = 1; m < 16; m <<= 1) mx = fmaxf(mx, __shfl_xor(mx, m, 32));
    float e = expf(s - mx);
    float sm = e;
    for (int m = 1; m < 16; m <<= 1) sm += __shfl_xor(sm, m, 32);
    int M = mrow + r, N = lane & 15;
    S[((size_t)(b * HEADS + h)) * 256 + M * 16 + N] = e / sm;
  }
}

// ---------------------------------------------------------------------------
// O = S (16x16) x V (16xHW): one WMMA per 16-pixel tile (K padded 16->32).
// ---------------------------------------------------------------------------
__global__ void __launch_bounds__(256) attn_apply(
    const float* __restrict__ S,
    const float* __restrict__ vb, int vCt, int vCo,
    float* __restrict__ out, int outCt, int outCo,
    int B, int HW) {
  int wave = (blockIdx.x * blockDim.x + threadIdx.x) >> 5;
  int lane = threadIdx.x & 31;
  int Nt = HW >> 4;
  if (wave >= B * HEADS * Nt) return;
  int nt = wave & (Nt - 1);
  int t0 = wave / Nt;
  int h = t0 % HEADS, b = t0 / HEADS;
  int n = (nt << 4) + (lane & 15);
  const float* Sb = S + ((size_t)(b * HEADS + h)) * 256;
  v16h af, bf;
#pragma unroll
  for (int i = 0; i < 16; i++) {
    int d = kmapA(lane, i);
    af[i] = (_Float16)((d < 16) ? Sb[(lane & 15) * 16 + d] : 0.f);
    int db = kmapA(lane, i);
    bf[i] = (_Float16)((db < 16)
              ? vb[((size_t)b * vCt + vCo + h * 16 + db) * HW + n] : 0.f);
  }
  v8f acc = {};
  acc = __builtin_amdgcn_wmma_f32_16x16x32_f16(false, af, false, bf,
                                               (short)0, acc, false, false);
  int mrow = (lane & 16) ? 8 : 0;
#pragma unroll
  for (int r = 0; r < 8; r++) {
    int c = h * 16 + mrow + r;
    out[((size_t)b * outCt + outCo + c) * HW + n] = acc[r];
  }
}

// ---------------------------------------------------------------------------
// gelu(y1) * y2  (exact erf gelu), y1/y2 are channel slices of same buffer
// ---------------------------------------------------------------------------
__global__ void __launch_bounds__(256) gelu_gate(
    const float* __restrict__ in, int inCt, int o1, int o2,
    float* __restrict__ out, int outCt, int oo,
    int B, int C, int HW) {
  int idx = blockIdx.x * blockDim.x + threadIdx.x;
  if (idx >= B * C * HW) return;
  int n = idx & (HW - 1);
  int c = (idx / HW) % C;
  int b = idx / (HW * C);
  float x = in[((size_t)b * inCt + o1 + c) * HW + n];
  float g = 0.5f * x * (1.f + erff(x * 0.70710678118654752f));
  float y = in[((size_t)b * inCt + o2 + c) * HW + n];
  out[((size_t)b * outCt + oo + c) * HW + n] = g * y;
}

// ---------------------------------------------------------------------------
// pixel_shuffle(r=2) + lrelu, writes into a channel slice of dst
// ---------------------------------------------------------------------------
__global__ void __launch_bounds__(256) pshuf_lrelu(
    const float* __restrict__ in, int inCt, int inCo,
    float* __restrict__ out, int outCt, int outCo,
    int B, int Cq, int H, int W) {
  int Ho = 2 * H, Wo = 2 * W;
  int HWo = Ho * Wo;
  int idx = blockIdx.x * blockDim.x + threadIdx.x;
  if (idx >= B * Cq * HWo) return;
  int n = idx & (HWo - 1);
  int c = (idx / HWo) % Cq;
  int b = idx / (HWo * Cq);
  int y = n / Wo, x = n & (Wo - 1);
  int r1 = y & 1, r2 = x & 1;
  float v = in[((size_t)b * inCt + inCo + c * 4 + r1 * 2 + r2) * (H * W)
               + (y >> 1) * W + (x >> 1)];
  v = (v >= 0.f) ? v : 0.1f * v;
  out[((size_t)b * outCt + outCo + c) * HWo + n] = v;
}

__global__ void __launch_bounds__(256) copy_slice(
    const float* __restrict__ s, int sCt, int sCo,
    float* __restrict__ d, int dCt, int dCo, int B, int C, int HW) {
  int idx = blockIdx.x * blockDim.x + threadIdx.x;
  if (idx >= B * C * HW) return;
  int n = idx & (HW - 1);
  int c = (idx / HW) % C;
  int b = idx / (HW * C);
  d[((size_t)b * dCt + dCo + c) * HW + n] = s[((size_t)b * sCt + sCo + c) * HW + n];
}

__global__ void __launch_bounds__(256) add_into(
    const float* __restrict__ s, int sCt, int sCo,
    float* __restrict__ d, int dCt, int dCo, int B, int C, int HW) {
  int idx = blockIdx.x * blockDim.x + threadIdx.x;
  if (idx >= B * C * HW) return;
  int n = idx & (HW - 1);
  int c = (idx / HW) % C;
  int b = idx / (HW * C);
  d[((size_t)b * dCt + dCo + c) * HW + n] += s[((size_t)b * sCt + sCo + c) * HW + n];
}

// ---------------------------------------------------------------------------
// Host orchestration
// ---------------------------------------------------------------------------
namespace {
struct PackedConv { const _Float16* w; int Cin, Cout, K, Kp, Mp, k3; };
static inline int ceilDiv(int a, int b) { return (a + b - 1) / b; }
static inline int ilog2(int v) { int s = 0; while ((1 << s) < v) s++; return s; }
static inline int pow2shift(int v) { return (v & (v - 1)) == 0 ? ilog2(v) : -1; }
}

extern "C" void kernel_launch(void* const* d_in, const int* in_sizes, int n_in,
                              void* d_out, int out_size, void* d_ws, size_t ws_size,
                              hipStream_t stream) {
  (void)in_sizes; (void)n_in; (void)out_size; (void)ws_size;
  const int B = 2;
  const int HWs = 64 * 64;     // 64x64 feature resolution

  auto F = [&](int idx) { return (const float*)d_in[idx]; };

  // -------- inputs (setup_inputs() dict insertion order, flattened) --------
  const float* x1 = F(0);
  const float* x2 = F(1);
  int ip = 2;
  const float* cf1_w = F(ip++); const float* cf1_b = F(ip++);
  const float* cf2_w = F(ip++); const float* cf2_b = F(ip++);
  const float* cf3_w = F(ip++); const float* cf3_b = F(ip++);
  // t1
  const float* t1_n1w = F(ip++); const float* t1_n1b = F(ip++);
  const float* t1_qkvw = F(ip++); const float* t1_dww = F(ip++);
  const float* t1_projw = F(ip++); const float* t1_temp = F(ip++);
  const float* t1_n2w = F(ip++); const float* t1_n2b = F(ip++);
  const float* t1_finw = F(ip++); const float* t1_fdww = F(ip++);
  const float* t1_foutw = F(ip++);
  // t2
  const float* t2_n1w = F(ip++); const float* t2_n1b = F(ip++);
  const float* t2_qw = F(ip++); const float* t2_qdww = F(ip++);
  const float* t2_kvw = F(ip++); const float* t2_kvdww = F(ip++);
  const float* t2_projw = F(ip++); const float* t2_temp = F(ip++);
  const float* t2_n2w = F(ip++); const float* t2_n2b = F(ip++);
  const float* t2_finw = F(ip++); const float* t2_fdww = F(ip++);
  const float* t2_foutw = F(ip++);
  // resblocks
  const float *rb1w[5], *rb1b[5], *rb2w[5], *rb2b[5];
  for (int r = 0; r < 5; r++) {
    rb1w[r] = F(ip++); rb1b[r] = F(ip++); rb2w[r] = F(ip++); rb2b[r] = F(ip++);
  }
  // ill1, ill2: up1_w, up1_b, up2_w, up2_b, hr_w, hr_b, last_w, last_b
  const float* ill[2][8];
  for (int p = 0; p < 2; p++)
    for (int j = 0; j < 8; j++) ill[p][j] = F(ip++);

  // -------- workspace bump allocator --------
  char* base = (char*)d_ws;
  size_t off = 0;
  auto allocB = [&](size_t bytes) -> void* {
    void* p = base + off;
    off += (bytes + 255) & ~(size_t)255;
    return p;
  };
  auto allocF = [&](size_t n) { return (float*)allocB(n * 4); };
  auto allocH = [&](size_t n) { return (_Float16*)allocB(n * 2); };

  // -------- weight packing --------
  auto packConv = [&](const float* w, int Cout, int K, int k3) -> PackedConv {
    PackedConv pc;
    pc.Cout = Cout; pc.K = K; pc.k3 = k3;
    pc.Kp = ceilDiv(K, 32); pc.Mp = ceilDiv(Cout, 16);
    pc.Cin = k3 ? K / 9 : K;
    _Float16* pk = allocH((size_t)pc.Kp * pc.Mp * 512);
    int total = pc.Kp * pc.Mp * 32;
    pack_w<<<ceilDiv(total, 256), 256, 0, stream>>>(w, pk, Cout, K, pc.Kp, pc.Mp,
                                                    pc.Cin, k3);
    pc.w = pk;
    return pc;
  };

  PackedConv pc_cf1 = packConv(cf1_w, 64, 27, 1);
  PackedConv pc_cf2 = packConv(cf2_w, 64, 576, 1);
  PackedConv pc_cf3 = packConv(cf3_w, 64, 576, 1);
  PackedConv pc_t1_qkv  = packConv(t1_qkvw, 3 * NF, NF, 0);
  PackedConv pc_t1_proj = packConv(t1_projw, NF, NF, 0);
  PackedConv pc_t1_fin  = packConv(t1_finw, 2 * HIDC, NF, 0);
  PackedConv pc_t1_fout = packConv(t1_foutw, NF, HIDC, 0);
  PackedConv pc_t2_q    = packConv(t2_qw, NF, NF, 0);
  PackedConv pc_t2_kv   = packConv(t2_kvw, 2 * NF, NF, 0);
  PackedConv pc_t2_proj = packConv(t2_projw, NF, NF, 0);
  PackedConv pc_t2_fin  = packConv(t2_finw, 2 * HIDC, NF, 0);
  PackedConv pc_t2_fout = packConv(t2_foutw, NF, HIDC, 0);
  PackedConv pc_rb1[5], pc_rb2[5];
  for (int r = 0; r < 5; r++) {
    pc_rb1[r] = packConv(rb1w[r], 128, 1152, 1);
    pc_rb2[r] = packConv(rb2w[r], 128, 1152, 1);
  }
  PackedConv pc_up1[2], pc_up2[2], pc_hr[2], pc_last[2];
  for (int p = 0; p < 2; p++) {
    pc_up1[p]  = packConv(ill[p][0], 256, 1152, 1);
    pc_up2[p]  = packConv(ill[p][2], 256, 1152, 1);
    pc_hr[p]   = packConv(ill[p][4], 64, 1152, 1);
    pc_last[p] = packConv(ill[p][6], 3, 576, 1);
  }

  // -------- activations --------
  float* a1 = allocF((size_t)B * 64 * 65536);
  float* a2 = allocF((size_t)B * 64 * 16384);
  float* a3 = allocF((size_t)B * 64 * 4096);
  float* b1 = allocF((size_t)B * 64 * 65536);
  float* b2 = allocF((size_t)B * 64 * 16384);
  float* b3 = allocF((size_t)B * 64 * 4096);
  float* L1buf = allocF((size_t)B * 64 * 4096);
  float* feaA = allocF((size_t)B * 128 * 4096);
  float* feaB = allocF((size_t)B * 128 * 4096);
  float* tbTmp = allocF((size_t)B * 64 * 4096);
  // transformer-block scratch (reused by all 4 block evaluations)
  float* s_ln = allocF((size_t)B * 64 * HWs);
  float* s_c  = allocF((size_t)B * 340 * HWs);
  float* s_d  = allocF((size_t)B * 340 * HWs);
  float* s_g  = allocF((size_t)B * HIDC * HWs);
  float* s_x1 = allocF((size_t)B * 64 * HWs);
  float* s_o  = allocF((size_t)B * 64 * HWs);
  float* s_S  = allocF((size_t)B * HEADS * 256);
  // up-path scratch
  float* cat1 = allocF((size_t)B * 128 * 4096);
  float* t64  = allocF((size_t)B * 256 * 4096);
  float* cat2 = allocF((size_t)B * 128 * 16384);
  float* t128 = allocF((size_t)B * 256 * 16384);
  float* cat3 = allocF((size_t)B * 128 * 65536);
  float* hrb  = allocF((size_t)B * 64 * 65536);

  // -------- launch helpers --------
  auto conv = [&](const float* in, int inCt, int inCo, const PackedConv& pc,
                  const float* bias, const float* resid, int resCt, int resCo,
                  float* out, int outCt, int outCo,
                  int Hin, int Win, int stride, int act) {
    int Hout = pc.k3 ? Hin / stride : Hin;
    int Wout = pc.k3 ? Win / stride : Win;
    int HWo = Hout * Wout;                 // always a power of two here
    int wsh = ilog2(Wout);
    int Ng = HWo >> 6;
    int gsh = ilog2(Ng);
    int csh = pow2shift(pc.Cin);
    int waves = B * pc.Mp * Ng;            // multiple of 8 (Ng >= 64)
    int blocks = ceilDiv(waves * 32, 256);
    if (pc.k3)
      conv_wmma<1><<<blocks, 256, 0, stream>>>(
          in, inCt, inCo, pc.w, bias, resid, resCt, resCo, out, outCt, outCo,
          B, pc.Cin, pc.Cout, Hin, Win, stride, act, pc.Mp, pc.Kp,
          wsh, HWo, gsh, csh);
    else
      conv_wmma<0><<<blocks, 256, 0, stream>>>(
          in, inCt, inCo, pc.w, bias, resid, resCt, resCo, out, outCt, outCo,
          B, pc.Cin, pc.Cout, Hin, Win, stride, act, pc.Mp, pc.Kp,
          wsh, HWo, gsh, csh);
  };
  auto chanLN = [&](const float* in, int iCt, int iCo, const float* w,
                    const float* bb, float* out, int oCt, int oCo, int C, int HW) {
    chan_ln_k<<<ceilDiv(B * HW, 256), 256, 0, stream>>>(
        in, iCt, iCo, w, bb, out, oCt, oCo, B, C, HW);
  };
  auto dw3 = [&](const float* in, int iCt, int iCo, const float* w,
                 float* out, int oCt, int oCo, int C, int H, int W) {
    dwconv3x3<<<ceilDiv(B * C * H * W, 256), 256, 0, stream>>>(
        in, iCt, iCo, w, out, oCt, oCo, B, C, H, W);
  };
  auto l2n = [&](float* buf, int Ct, int Co, int C, int HW) {
    l2norm_rows<<<B * C, 256, 0, stream>>>(buf, Ct, Co, B, C, HW);
  };
  auto copySl = [&](const float* s, int sCt, int sCo, float* d, int dCt,
                    int dCo, int C, int HW) {
    copy_slice<<<ceilDiv(B * C * HW, 256), 256, 0, stream>>>(
        s, sCt, sCo, d, dCt, dCo, B, C, HW);
  };
  auto addSl = [&](const float* s, int sCt, int sCo, float* d, int dCt,
                   int dCo, int C, int HW) {
    add_into<<<ceilDiv(B * C * HW, 256), 256, 0, stream>>>(
        s, sCt, sCo, d, dCt, dCo, B, C, HW);
  };

  // attention core: q/k/v staged in s_d at channel offsets 0/64/128 (Ct=192)
  auto attnCore = [&](const float* temp, const PackedConv& proj,
                      const float* rx, int rCt, int rCo,
                      float* out, int oCt, int oCo) {
    l2n(s_d, 192, 0, 64, HWs);     // q
    l2n(s_d, 192, 64, 64, HWs);    // k
    attn_gram<<<1, 256, 0, stream>>>(s_d, 192, 0, s_d, 192, 64, temp, s_S, B, HWs);
    int waves = B * HEADS * (HWs >> 4);
    attn_apply<<<ceilDiv(waves * 32, 256), 256, 0, stream>>>(
        s_S, s_d, 192, 128, s_o, 64, 0, B, HWs);
    conv(s_o, 64, 0, proj, nullptr, rx, rCt, rCo, out, oCt, oCo, 64, 64, 1, ACT_NONE);
  };

  auto ffn = [&](const float* n2w, const float* n2b, const PackedConv& fin,
                 const float* fdw, const PackedConv& fout,
                 float* out, int oCt, int oCo) {
    // input is s_x1 (x after attention residual)
    chanLN(s_x1, 64, 0, n2w, n2b, s_ln, 64, 0, 64, HWs);
    conv(s_ln, 64, 0, fin, nullptr, nullptr, 0, 0, s_c, 340, 0, 64, 64, 1, ACT_NONE);
    dw3(s_c, 340, 0, fdw, s_d, 340, 0, 340, 64, 64);
    gelu_gate<<<ceilDiv(B * HIDC * HWs, 256), 256, 0, stream>>>(
        s_d, 340, 0, HIDC, s_g, HIDC, 0, B, HIDC, HWs);
    conv(s_g, HIDC, 0, fout, nullptr, s_x1, 64, 0, out, oCt, oCo, 64, 64, 1, ACT_NONE);
  };

  auto tblock = [&](const float* x, int xCt, int xCo,
                    float* out, int oCt, int oCo) {
    chanLN(x, xCt, xCo, t1_n1w, t1_n1b, s_ln, 64, 0, 64, HWs);
    conv(s_ln, 64, 0, pc_t1_qkv, nullptr, nullptr, 0, 0, s_c, 192, 0, 64, 64, 1, ACT_NONE);
    dw3(s_c, 192, 0, t1_dww, s_d, 192, 0, 192, 64, 64);
    attnCore(t1_temp, pc_t1_proj, x, xCt, xCo, s_x1, 64, 0);
    ffn(t1_n2w, t1_n2b, pc_t1_fin, t1_fdww, pc_t1_fout, out, oCt, oCo);
  };

  auto tblock2 = [&](const float* x, int xCt, int xCo,
                     const float* xx, int xxCt, int xxCo,
                     float* out, int oCt, int oCo) {
    chanLN(x, xCt, xCo, t2_n1w, t2_n1b, s_ln, 64, 0, 64, HWs);
    conv(s_ln, 64, 0, pc_t2_q, nullptr, nullptr, 0, 0, s_c, 192, 0, 64, 64, 1, ACT_NONE);
    chanLN(xx, xxCt, xxCo, t2_n1w, t2_n1b, s_ln, 64, 0, 64, HWs);
    conv(s_ln, 64, 0, pc_t2_kv, nullptr, nullptr, 0, 0, s_c, 192, 64, 64, 64, 1, ACT_NONE);
    dw3(s_c, 192, 0, t2_qdww, s_d, 192, 0, 64, 64, 64);
    dw3(s_c, 192, 64, t2_kvdww, s_d, 192, 64, 128, 64, 64);
    attnCore(t2_temp, pc_t2_proj, x, xCt, xCo, s_x1, 64, 0);
    ffn(t2_n2w, t2_n2b, pc_t2_fin, t2_fdww, pc_t2_fout, out, oCt, oCo);
  };

  // -------- first(x): 3->64 @256^2, s2 -> 128^2, s2 -> 64^2 (lrelu) --------
  conv(x1, 3, 0, pc_cf1, cf1_b, nullptr, 0, 0, a1, 64, 0, 256, 256, 1, ACT_LRELU);
  conv(a1, 64, 0, pc_cf2, cf2_b, nullptr, 0, 0, a2, 64, 0, 256, 256, 2, ACT_LRELU);
  conv(a2, 64, 0, pc_cf3, cf3_b, nullptr, 0, 0, a3, 64, 0, 128, 128, 2, ACT_LRELU);
  conv(x2, 3, 0, pc_cf1, cf1_b, nullptr, 0, 0, b1, 64, 0, 256, 256, 1, ACT_LRELU);
  conv(b1, 64, 0, pc_cf2, cf2_b, nullptr, 0, 0, b2, 64, 0, 256, 256, 2, ACT_LRELU);
  conv(b2, 64, 0, pc_cf3, cf3_b, nullptr, 0, 0, b3, 64, 0, 128, 128, 2, ACT_LRELU);

  // -------- L1 = tblock(a3,t1) + tblock2(a3,b3,t2) --------
  tblock(a3, 64, 0, L1buf, 64, 0);
  tblock2(a3, 64, 0, b3, 64, 0, tbTmp, 64, 0);
  addSl(tbTmp, 64, 0, L1buf, 64, 0, 64, HWs);
  // -------- L2 = tblock(b3,t1) + tblock2(b3,L1,t2) -> feaA[:,64:128] ------
  tblock(b3, 64, 0, feaA, 128, 64);
  tblock2(b3, 64, 0, L1buf, 64, 0, tbTmp, 64, 0);
  addSl(tbTmp, 64, 0, feaA, 128, 64, 64, HWs);
  // fea = concat(L1, L2)
  copySl(L1buf, 64, 0, feaA, 128, 0, 64, HWs);

  // -------- resblocks (ping-pong), relu-conv fused --------
  float* cur = feaA;
  float* nxt = feaB;
  for (int r = 0; r < 5; r++) {
    conv(cur, 128, 0, pc_rb1[r], rb1b[r], nullptr, 0, 0, s_c, 128, 0, 64, 64, 1, ACT_RELU);
    conv(s_c, 128, 0, pc_rb2[r], rb2b[r], cur, 128, 0, nxt, 128, 0, 64, 64, 1, ACT_NONE);
    float* t = cur; cur = nxt; nxt = t;
  }

  // -------- up paths (shared concat staging) --------
  copySl(cur, 128, 0, cat1, 128, 0, 64, 4096);    // out_fea1 = fea[:, :64]
  copySl(L1buf, 64, 0, cat1, 128, 64, 64, 4096);  // f3 = L1
  copySl(a2, 64, 0, cat2, 128, 64, 64, 16384);    // f2 = a2
  copySl(a1, 64, 0, cat3, 128, 64, 64, 65536);    // f1 = a1

  float* outp = (float*)d_out;  // (2, 6, 256, 256): o1 -> ch 0..2, o2 -> ch 3..5
  for (int p = 0; p < 2; p++) {
    conv(cat1, 128, 0, pc_up1[p], ill[p][1], nullptr, 0, 0, t64, 256, 0, 64, 64, 1, ACT_NONE);
    pshuf_lrelu<<<ceilDiv(B * 64 * 16384, 256), 256, 0, stream>>>(
        t64, 256, 0, cat2, 128, 0, B, 64, 64, 64);
    conv(cat2, 128, 0, pc_up2[p], ill[p][3], nullptr, 0, 0, t128, 256, 0, 128, 128, 1, ACT_NONE);
    pshuf_lrelu<<<ceilDiv(B * 64 * 65536, 256), 256, 0, stream>>>(
        t128, 256, 0, cat3, 128, 0, B, 64, 128, 128);
    conv(cat3, 128, 0, pc_hr[p], ill[p][5], nullptr, 0, 0, hrb, 64, 0, 256, 256, 1, ACT_LRELU);
    conv(hrb, 64, 0, pc_last[p], ill[p][7], nullptr, 0, 0, outp, 6, p * 3, 256, 256, 1, ACT_SIGMOID);
  }
}
